// AttnBlock_6382321402253
// MI455X (gfx1250) — matrix operand order repaired and checked by reading, not compiled, not run on hardware
//
#include <hip/hip_runtime.h>
#include <hip/hip_bf16.h>

#define B_   2
#define C_   512
#define N_   4096
#define G_   32
#define CG_  16
#define EPS_ 1e-6f

typedef __bf16 bf16;
typedef __attribute__((ext_vector_type(16))) __bf16 v16bf;
typedef __attribute__((ext_vector_type(8)))  float  v8f;

static __device__ __forceinline__ v8f wmma_bf16(v16bf a, v16bf b, v8f c) {
    return __builtin_amdgcn_wmma_f32_16x16x32_bf16(
        /*neg_a=*/false, a, /*neg_b=*/false, b,
        /*c_mod=*/(short)0, c, /*reuse_a=*/false, /*reuse_b=*/false);
}

// A-matrix 16x32 bf16 fragment from row-major src (ld = K stride), per ISA 7.12.2:
// lane m=lane&15 holds row m; VGPR r holds K = (r/4)*16 + (lane/16)*8 + (r%4)*2 (+1)
static __device__ __forceinline__ v16bf load_A(const bf16* a, int ld, int lane) {
    int m  = lane & 15;
    int kh = ((lane >> 4) & 1) << 3;
    const bf16* row = a + (size_t)m * ld;
    v16bf f;
#pragma unroll
    for (int r = 0; r < 8; ++r) {
        int k = ((r >> 2) << 4) + kh + ((r & 3) << 1);
        f[2 * r]     = row[k];
        f[2 * r + 1] = row[k + 1];
    }
    return f;
}

// B-matrix 32x16 bf16 fragment from "NT" storage: b[col][k] row-major (ld = K stride).
// lane col=lane&15; element s holds K = 8h+s (s<8), 16+8h+(s-8) (s>=8), h=lane/16 (matches load_A).
static __device__ __forceinline__ v16bf load_B(const bf16* b, int ld, int lane) {
    const bf16* col = b + (size_t)(lane & 15) * ld + (((lane >> 4) & 1) << 3);
    v16bf f;
#pragma unroll
    for (int r = 0; r < 16; ++r) f[r] = col[r + ((r >> 3) << 3)];
    return f;
}

// ---------------------------------------------------------------- weights -> bf16
__global__ void cast_w_kernel(const float* __restrict__ qw, const float* __restrict__ kw,
                              const float* __restrict__ vw, const float* __restrict__ pw,
                              bf16* __restrict__ out) {
    int i = blockIdx.x * blockDim.x + threadIdx.x;  // 0 .. 512*512-1
    const int W2 = C_ * C_;
    out[i]          = (bf16)qw[i];
    out[W2 + i]     = (bf16)kw[i];
    out[2 * W2 + i] = (bf16)vw[i];
    out[3 * W2 + i] = (bf16)pw[i];
}

// ---------------------------------------------------------------- groupnorm stats
__global__ void gn_stats_kernel(const float* __restrict__ x, float* __restrict__ stats) {
    int bg = blockIdx.x;          // 0..63
    int b = bg / G_, g = bg % G_;
    const float4* base = (const float4*)(x + (size_t)b * C_ * N_ + (size_t)g * CG_ * N_);
    const int TOT4 = (CG_ * N_) / 4;   // 16384
    int tid = threadIdx.x;
    float s = 0.f, ss = 0.f;
    for (int i = tid; i < TOT4; i += 256) {
        float4 v = base[i];
        s  += v.x + v.y + v.z + v.w;
        ss += v.x * v.x + v.y * v.y + v.z * v.z + v.w * v.w;
    }
    __shared__ float rs[256], rss[256];
    rs[tid] = s; rss[tid] = ss;
    __syncthreads();
    for (int off = 128; off > 0; off >>= 1) {
        if (tid < off) { rs[tid] += rs[tid + off]; rss[tid] += rss[tid + off]; }
        __syncthreads();
    }
    if (tid == 0) {
        const float inv = 1.0f / (float)(CG_ * N_);
        float mean = rs[0] * inv;
        float var  = rss[0] * inv - mean * mean;
        stats[2 * bg]     = mean;
        stats[2 * bg + 1] = rsqrtf(var + EPS_);
    }
}

// ------------------------------------------- groupnorm apply + transpose -> hT[b][n][c] bf16
__global__ void gn_apply_kernel(const float* __restrict__ x, const float* __restrict__ gw,
                                const float* __restrict__ gb, const float* __restrict__ stats,
                                bf16* __restrict__ hT) {
    int b  = blockIdx.z;
    int n0 = blockIdx.x * 32, c0 = blockIdx.y * 32;
    __shared__ float tile[32][33];
    int tx = threadIdx.x, ty = threadIdx.y;
#pragma unroll
    for (int r = 0; r < 4; ++r) {
        int cl = ty + 8 * r;
        int c  = c0 + cl;
        int g  = c >> 4;
        float mu  = stats[2 * (b * G_ + g)];
        float rsg = stats[2 * (b * G_ + g) + 1];
        float v = x[(size_t)b * C_ * N_ + (size_t)c * N_ + n0 + tx];
        tile[cl][tx] = (v - mu) * rsg * gw[c] + gb[c];
    }
    __syncthreads();
#pragma unroll
    for (int r = 0; r < 4; ++r) {
        int nl = ty + 8 * r;
        hT[(size_t)b * N_ * C_ + (size_t)(n0 + nl) * C_ + c0 + tx] = (bf16)tile[tx][nl];
    }
}

// ---------------------------------------------------------------- generic NT GEMM (WMMA)
// D[m][n] = sum_k A[m][k] * B[n][k]  (+bias, +residual per MODE)
// Block tile 128x128, wave tile 32x64 (acc[2][4]): 6 fragment loads feed 8 WMMAs.
// MODE 0: bf16 out, bias indexed by col ; MODE 1: bf16 out, bias by row
// MODE 2: f32 out, bias by row, + residual
template <int MODE>
__global__ void __launch_bounds__(256) gemm_nt_kernel(
        const bf16* __restrict__ A, long sAb, const bf16* __restrict__ Bm, long sBb,
        const float* __restrict__ bias, bf16* __restrict__ outb, float* __restrict__ outf,
        long sOb, const float* __restrict__ resid, long sRb, int M, int NC, int K) {
    int bb = blockIdx.z;
    A  += (size_t)bb * sAb;
    Bm += (size_t)bb * sBb;
    int m0 = blockIdx.x * 128, n0 = blockIdx.y * 128;
    int wave = threadIdx.x >> 5, lane = threadIdx.x & 31;
    int wm0 = m0 + (wave >> 1) * 32;
    int wn0 = n0 + (wave & 1) * 64;
    v8f acc[2][4] = {};
    for (int k0 = 0; k0 < K; k0 += 32) {
        v16bf a0 = load_A(A + (size_t)wm0 * K + k0, K, lane);
        v16bf a1 = load_A(A + (size_t)(wm0 + 16) * K + k0, K, lane);
#pragma unroll
        for (int tn = 0; tn < 4; ++tn) {
            v16bf b = load_B(Bm + (size_t)(wn0 + tn * 16) * K + k0, K, lane);
            acc[0][tn] = wmma_bf16(a0, b, acc[0][tn]);
            acc[1][tn] = wmma_bf16(a1, b, acc[1][tn]);
        }
    }
    int cn = lane & 15, hl = (lane >> 4) & 1;
#pragma unroll
    for (int tm = 0; tm < 2; ++tm)
#pragma unroll
        for (int tn = 0; tn < 4; ++tn) {
            int colg = wn0 + tn * 16 + cn;
#pragma unroll
            for (int r = 0; r < 8; ++r) {
                int rowg = wm0 + tm * 16 + hl * 8 + r;
                float vo = acc[tm][tn][r];
                vo += (MODE == 0) ? bias[colg] : bias[rowg];
                size_t idx = (size_t)rowg * NC + colg;
                if (MODE == 2)
                    outf[(size_t)bb * sOb + idx] = vo + resid[(size_t)bb * sRb + idx];
                else
                    outb[(size_t)bb * sOb + idx] = (bf16)vo;
            }
        }
}

// ---------------------------------------------------------------- flash attention
// qT,kT: [B][N][C] bf16 ; v: [B][C][N] bf16 ; attnT out: [B][N][C] bf16
// Block = 32 query rows (two 16-row tiles), 8 waves; wave owns a 16-col j-slice for S
// and a 64-wide channel slice for P*V. K/V fragments are each reused by 2 WMMAs.
__global__ void __launch_bounds__(256) attn_kernel(const bf16* __restrict__ qT,
                                                   const bf16* __restrict__ kT,
                                                   const bf16* __restrict__ vM,
                                                   bf16* __restrict__ attnT) {
    const float scale = 0.044194173824159216f;  // 512^-0.5
    int bb = blockIdx.y;
    int i0 = blockIdx.x * 32;
    int wave = threadIdx.x >> 5, lane = threadIdx.x & 31;
    int hl = (lane >> 4) & 1;
    const bf16* q  = qT + (size_t)bb * N_ * C_ + (size_t)i0 * C_;
    const bf16* kk = kT + (size_t)bb * N_ * C_;
    const bf16* vv = vM + (size_t)bb * C_ * N_;

    __shared__ bf16  sP[32 * 128];
    __shared__ float sMax[8][32];
    __shared__ float sSum[8][32];

    // Preload all Q fragments for this 32-row tile (2 x 16 fragments, reused 32x)
    v16bf qf[2][16];
#pragma unroll
    for (int rt = 0; rt < 2; ++rt)
#pragma unroll
        for (int t = 0; t < 16; ++t)
            qf[rt][t] = load_A(q + (size_t)(rt * 16) * C_ + t * 32, C_, lane);

    v8f O[2][4] = {};
    float mrun[2][8], lrun[2][8];
#pragma unroll
    for (int rt = 0; rt < 2; ++rt)
#pragma unroll
        for (int r = 0; r < 8; ++r) { mrun[rt][r] = -1e30f; lrun[rt][r] = 0.f; }

    int c0 = wave * 64;  // this wave's output-channel slice for P*V

    for (int step = 0; step < N_ / 128; ++step) {
        int j0 = step * 128 + wave * 16;
        if (step < N_ / 128 - 1) {  // prefetch next step's K/V fragment bases into caches
            __builtin_prefetch(kk + (size_t)(j0 + 128) * C_ + (size_t)(lane & 15) * C_ +
                                   (((lane >> 4) & 1) << 4), 0, 3);
            __builtin_prefetch(vv + (size_t)(c0 + (lane & 15)) * N_ + (step + 1) * 128 +
                                   (((lane >> 4) & 1) << 4), 0, 3);
        }
        // ---- S tiles: 2 x (16 rows x 16 cols); one K fragment feeds both row-tiles
        v8f S[2] = {};
#pragma unroll
        for (int t = 0; t < 16; ++t) {
            v16bf kf = load_B(kk + (size_t)j0 * C_ + t * 32, C_, lane);
            S[0] = wmma_bf16(qf[0][t], kf, S[0]);
            S[1] = wmma_bf16(qf[1][t], kf, S[1]);
        }
        // ---- row max within wave (16 lanes hold the 16 j-cols of a row)
#pragma unroll
        for (int rt = 0; rt < 2; ++rt) {
            float rmax[8];
#pragma unroll
            for (int r = 0; r < 8; ++r) {
                float sv = S[rt][r] * scale;
                S[rt][r] = sv;
                float mx = sv;
                mx = fmaxf(mx, __shfl_xor(mx, 1, 32));
                mx = fmaxf(mx, __shfl_xor(mx, 2, 32));
                mx = fmaxf(mx, __shfl_xor(mx, 4, 32));
                mx = fmaxf(mx, __shfl_xor(mx, 8, 32));
                rmax[r] = mx;
            }
            if ((lane & 15) == 0) {
#pragma unroll
                for (int r = 0; r < 8; ++r) sMax[wave][rt * 16 + hl * 8 + r] = rmax[r];
            }
        }
        __syncthreads();
        // ---- combine max across the 8 waves (full 128-col step), online rescale
        float corr[2][8];
#pragma unroll
        for (int rt = 0; rt < 2; ++rt)
#pragma unroll
            for (int r = 0; r < 8; ++r) {
                float mx = mrun[rt][r];
#pragma unroll
                for (int w2 = 0; w2 < 8; ++w2)
                    mx = fmaxf(mx, sMax[w2][rt * 16 + hl * 8 + r]);
                corr[rt][r] = __expf(mrun[rt][r] - mx);
                mrun[rt][r] = mx;
            }
        // ---- P = exp(S - m), wave-local row sums, O rescale, stage P to LDS (bf16)
#pragma unroll
        for (int rt = 0; rt < 2; ++rt) {
            float rsum[8];
#pragma unroll
            for (int r = 0; r < 8; ++r) {
                float p = __expf(S[rt][r] - mrun[rt][r]);
                float su = p;
                su += __shfl_xor(su, 1, 32);
                su += __shfl_xor(su, 2, 32);
                su += __shfl_xor(su, 4, 32);
                su += __shfl_xor(su, 8, 32);
                rsum[r] = su;
                O[rt][0][r] *= corr[rt][r];
                O[rt][1][r] *= corr[rt][r];
                O[rt][2][r] *= corr[rt][r];
                O[rt][3][r] *= corr[rt][r];
                sP[(rt * 16 + hl * 8 + r) * 128 + wave * 16 + (lane & 15)] = (bf16)p;
            }
            if ((lane & 15) == 0) {
#pragma unroll
                for (int r = 0; r < 8; ++r) sSum[wave][rt * 16 + hl * 8 + r] = rsum[r];
            }
        }
        __syncthreads();
#pragma unroll
        for (int rt = 0; rt < 2; ++rt)
#pragma unroll
            for (int r = 0; r < 8; ++r) {
                float su = 0.f;
#pragma unroll
                for (int w2 = 0; w2 < 8; ++w2) su += sSum[w2][rt * 16 + hl * 8 + r];
                lrun[rt][r] = lrun[rt][r] * corr[rt][r] + su;
            }
        // ---- O(i, c-slice) += P(32x128) * V(c,j)^T ; A from LDS, B from global v[C][N];
        //      one V fragment feeds both row-tiles.
#pragma unroll
        for (int t = 0; t < 4; ++t) {
            v16bf aP0 = load_A(sP + t * 32, 128, lane);
            v16bf aP1 = load_A(sP + 16 * 128 + t * 32, 128, lane);
#pragma unroll
            for (int ct = 0; ct < 4; ++ct) {
                v16bf bV = load_B(vv + (size_t)(c0 + ct * 16) * N_ + step * 128 + t * 32,
                                  N_, lane);
                O[0][ct] = wmma_bf16(aP0, bV, O[0][ct]);
                O[1][ct] = wmma_bf16(aP1, bV, O[1][ct]);
            }
        }
        __syncthreads();
    }
    // ---- normalize and write attnT[i][c]
    bf16* outp = attnT + (size_t)bb * N_ * C_ + (size_t)i0 * C_;
#pragma unroll
    for (int rt = 0; rt < 2; ++rt)
#pragma unroll
        for (int r = 0; r < 8; ++r) {
            float inv = 1.f / lrun[rt][r];
            int row = rt * 16 + hl * 8 + r;
#pragma unroll
            for (int ct = 0; ct < 4; ++ct)
                outp[(size_t)row * C_ + c0 + ct * 16 + (lane & 15)] =
                    (bf16)(O[rt][ct][r] * inv);
        }
}

// ---------------------------------------------------------------- host launcher
extern "C" void kernel_launch(void* const* d_in, const int* in_sizes, int n_in,
                              void* d_out, int out_size, void* d_ws, size_t ws_size,
                              hipStream_t stream) {
    const float* x   = (const float*)d_in[0];
    const float* gnw = (const float*)d_in[1];
    const float* gnb = (const float*)d_in[2];
    const float* qw  = (const float*)d_in[3];
    const float* qb  = (const float*)d_in[4];
    const float* kw  = (const float*)d_in[5];
    const float* kb  = (const float*)d_in[6];
    const float* vw  = (const float*)d_in[7];
    const float* vb  = (const float*)d_in[8];
    const float* pw  = (const float*)d_in[9];
    const float* pb  = (const float*)d_in[10];
    float* out = (float*)d_out;

    char* ws = (char*)d_ws;
    size_t off = 0;
    float* stats = (float*)(ws + off); off += 1024;                 // 64 x {mu, rsigma}
    bf16* wq = (bf16*)(ws + off); off += (size_t)C_ * C_ * 2;
    bf16* wk = (bf16*)(ws + off); off += (size_t)C_ * C_ * 2;
    bf16* wv = (bf16*)(ws + off); off += (size_t)C_ * C_ * 2;
    bf16* wp = (bf16*)(ws + off); off += (size_t)C_ * C_ * 2;
    bf16* hT    = (bf16*)(ws + off); off += (size_t)B_ * N_ * C_ * 2;
    bf16* qT    = (bf16*)(ws + off); off += (size_t)B_ * N_ * C_ * 2;
    bf16* kT    = (bf16*)(ws + off); off += (size_t)B_ * N_ * C_ * 2;
    bf16* vbuf  = (bf16*)(ws + off); off += (size_t)B_ * C_ * N_ * 2;
    bf16* attnT = (bf16*)(ws + off); off += (size_t)B_ * N_ * C_ * 2;
    (void)off; (void)ws_size; (void)in_sizes; (void)n_in; (void)out_size;

    const long sNC = (long)N_ * C_;   // batch stride of [N][C] buffers
    const long sCN = (long)C_ * N_;   // batch stride of [C][N] buffers

    cast_w_kernel<<<(C_ * C_) / 256, 256, 0, stream>>>(qw, kw, vw, pw, wq);
    gn_stats_kernel<<<B_ * G_, 256, 0, stream>>>(x, stats);
    gn_apply_kernel<<<dim3(N_ / 32, C_ / 32, B_), dim3(32, 8), 0, stream>>>(
        x, gnw, gnb, stats, hT);

    // qT[n][o] = sum_c hT[n][c] * qw[o][c] + qb[o]   (bias per col)
    gemm_nt_kernel<0><<<dim3(N_ / 128, C_ / 128, B_), 256, 0, stream>>>(
        hT, sNC, wq, 0L, qb, qT, nullptr, sNC, nullptr, 0L, N_, C_, C_);
    gemm_nt_kernel<0><<<dim3(N_ / 128, C_ / 128, B_), 256, 0, stream>>>(
        hT, sNC, wk, 0L, kb, kT, nullptr, sNC, nullptr, 0L, N_, C_, C_);
    // v[o][n] = sum_c vw[o][c] * hT[n][c] + vb[o]    (bias per row)
    gemm_nt_kernel<1><<<dim3(C_ / 128, N_ / 128, B_), 256, 0, stream>>>(
        wv, 0L, hT, sNC, vb, vbuf, nullptr, sCN, nullptr, 0L, C_, N_, C_);

    attn_kernel<<<dim3(N_ / 32, B_), 256, 0, stream>>>(qT, kT, vbuf, attnT);

    // out[o][n] = x[o][n] + sum_c pw[o][c] * attnT[n][c] + pb[o]
    gemm_nt_kernel<2><<<dim3(C_ / 128, N_ / 128, B_), 256, 0, stream>>>(
        wp, 0L, attnT, sNC, pb, nullptr, out, sCN, x, sCN, C_, N_, C_);
}
